// MHA_76519137346007
// MI455X (gfx1250) — matrix operand order repaired and checked by reading, not compiled, not run on hardware
//
#include <hip/hip_runtime.h>
#include <hip/hip_bf16.h>
#include <cstdint>

// ---------------------------------------------------------------------------
// MHA forward for B=2, N=2048, E=1024, H=16, D=64 on gfx1250 (CDNA5).
// f16 WMMA (v_wmma_f32_16x16x32_f16) with f32 accumulate.
// Flash-style streaming softmax; K/V staged in LDS via async-to-LDS DMA,
// double-buffered, shared by 8 waves per workgroup.
// ---------------------------------------------------------------------------

typedef _Float16 half8 __attribute__((ext_vector_type(8)));
typedef _Float16 v16h  __attribute__((ext_vector_type(16)));
typedef float    v8f   __attribute__((ext_vector_type(8)));

#define WMMA_F16(a, b, c) \
  __builtin_amdgcn_wmma_f32_16x16x32_f16(false, (a), false, (b), (short)0, (c), false, false)

// A-matrix fragment (16x32 f16): lane holds its row; halves 0-7 <- K = 8h..8h+7,
// halves 8-15 <- K = 16+8h..16+8h+7  (ISA 7.12.2, 16-bit A 16x32 table).
__device__ __forceinline__ v16h load_fragA(const _Float16* __restrict__ row, int koff, int h) {
  half8 lo = *(const half8*)(row + koff + 8 * h);
  half8 hi = *(const half8*)(row + koff + 16 + 8 * h);
  return __builtin_shufflevector(lo, hi, 0, 1, 2, 3, 4, 5, 6, 7, 8, 9, 10, 11, 12, 13, 14, 15);
}

// B-matrix fragment (32x16 f16): lane holds its column; halves 0-7 <- K = 8h..8h+7,
// halves 8-15 <- K = 16+8h..16+8h+7.
__device__ __forceinline__ v16h load_fragB(const _Float16* __restrict__ row, int koff, int h) {
  half8 lo = *(const half8*)(row + koff + 8 * h);
  half8 hi = *(const half8*)(row + koff + 16 + 8 * h);
  return __builtin_shufflevector(lo, hi, 0, 1, 2, 3, 4, 5, 6, 7, 8, 9, 10, 11, 12, 13, 14, 15);
}

// Async global->LDS 16B copy (CDNA5 GLOBAL_LOAD_ASYNC_TO_LDS_B128, ASYNCcnt).
// lds_addr: raw LDS byte address (= low 32 bits of the flat shared pointer).
__device__ __forceinline__ void async_copy_b128(uint32_t lds_addr, const void* gaddr) {
  asm volatile("global_load_async_to_lds_b128 %0, %1, off"
               :: "v"(lds_addr), "v"((uint64_t)(uintptr_t)gaddr)
               : "memory");
}
__device__ __forceinline__ void wait_async0() {
  asm volatile("s_wait_asynccnt 0" ::: "memory");
}

// ---------------------------------------------------------------------------
// f32 -> f16 conversion
// ---------------------------------------------------------------------------
__global__ void cvt_f32_f16(const float* __restrict__ src, _Float16* __restrict__ dst, int n) {
  int i = blockIdx.x * blockDim.x + threadIdx.x;
  if (i < n) dst[i] = (_Float16)src[i];
}

// ---------------------------------------------------------------------------
// GEMM: Y[m,n] = sum_k A[m,k] * W[n,k] + bias[n]. One wave: 32x64 output tile
// (2 A-frags reused across 4 B-frags -> 8 WMMA per 12 B128 loads per K-step).
// mode 0: f16 row-major [M,1024]        (Q, K projections)
// mode 1: f16 transposed V^T [B,H,D,N]  (V projection)
// mode 2: f32 row-major [M,1024]        (output projection -> d_out)
// ---------------------------------------------------------------------------
__global__ __launch_bounds__(128) void gemm_wmma(
    const _Float16* __restrict__ A, const _Float16* __restrict__ W,
    const float* __restrict__ bias, void* __restrict__ out,
    int M, int K, int mode) {
  const int lane = threadIdx.x & 31;
  const int wave = threadIdx.x >> 5;
  const int ln   = lane & 15;
  const int h    = lane >> 4;
  const int m0   = blockIdx.x * 32;
  const int c0   = (blockIdx.y * 4 + wave) * 64;

  const _Float16* arow0 = A + (long)(m0 + ln) * K;
  const _Float16* arow1 = A + (long)(m0 + 16 + ln) * K;
  v8f acc[2][4] = {};

  for (int kb = 0; kb < K; kb += 32) {
    v16h a0 = load_fragA(arow0, kb, h);
    v16h a1 = load_fragA(arow1, kb, h);
#pragma unroll
    for (int t = 0; t < 4; ++t) {
      const _Float16* wrow = W + (long)(c0 + t * 16 + ln) * K;
      v16h bf = load_fragB(wrow, kb, h);
      acc[0][t] = WMMA_F16(a0, bf, acc[0][t]);
      acc[1][t] = WMMA_F16(a1, bf, acc[1][t]);
    }
  }

  if (mode == 0) {                       // f16 row-major
    _Float16* O = (_Float16*)out;
#pragma unroll
    for (int r = 0; r < 2; ++r)
#pragma unroll
      for (int t = 0; t < 4; ++t) {
        const int n = c0 + t * 16 + ln;
        const float bv = bias[n];
#pragma unroll
        for (int v = 0; v < 8; ++v) {
          const int m = m0 + r * 16 + v + 8 * h;
          O[(long)m * 1024 + n] = (_Float16)(acc[r][t][v] + bv);
        }
      }
  } else if (mode == 1) {                // V^T : [B=2, H=16, D=64, N=2048] f16
    _Float16* O = (_Float16*)out;
    const int b = m0 >> 11;              // 2048 rows per batch; 32 | 2048 so uniform
#pragma unroll
    for (int r = 0; r < 2; ++r) {
      const int i0 = (m0 & 2047) + r * 16 + 8 * h;
#pragma unroll
      for (int t = 0; t < 4; ++t) {
        const int n  = c0 + t * 16 + ln;
        const int hh = n >> 6;
        const int d  = n & 63;
        const float bv = bias[n];
        half8 pk;
#pragma unroll
        for (int v = 0; v < 8; ++v) pk[v] = (_Float16)(acc[r][t][v] + bv);
        *(half8*)(O + (((long)(b * 16 + hh) * 64 + d) * 2048 + i0)) = pk;
      }
    }
  } else {                               // f32 row-major (final output)
    float* O = (float*)out;
#pragma unroll
    for (int r = 0; r < 2; ++r)
#pragma unroll
      for (int t = 0; t < 4; ++t) {
        const int n = c0 + t * 16 + ln;
        const float bv = bias[n];
#pragma unroll
        for (int v = 0; v < 8; ++v) {
          const int m = m0 + r * 16 + v + 8 * h;
          O[(long)m * 1024 + n] = acc[r][t][v] + bv;
        }
      }
  }
}

// ---------------------------------------------------------------------------
// Flash attention. Q,K: f16 [B,N,E] (head = 64-col slice). Vt: f16 [B,H,D,N].
// 256 threads = 8 waves per block; each wave owns 16 query rows. K/V blocks
// (32 keys) are staged in LDS by async DMA, double-buffered, shared by all
// 8 waves. T = K x Q^T (= S^T tile) puts each query in one lane: softmax is
// per-lane + one shfl_xor(16), and exp'd values map lane-locally into the
// P A-fragment for O += P x V.
// ---------------------------------------------------------------------------
__global__ __launch_bounds__(256) void attn_flash(
    const _Float16* __restrict__ Q, const _Float16* __restrict__ Kc,
    const _Float16* __restrict__ Vt, _Float16* __restrict__ O) {
  const int EE = 1024, NN = 2048;
  const float scale = 0.03125f;          // E^-0.5 = 1/32 (module scales by embed_size)

  // Per buffer: K tile [32 keys][64 d] + V tile [64 d][32 keys] = 8 KB.
  __shared__ __align__(16) _Float16 smem[2][32 * 64 + 64 * 32];

  const int tid  = threadIdx.x;
  const int lane = tid & 31;
  const int wave = tid >> 5;
  const int ln   = lane & 15;
  const int h    = lane >> 4;
  const int bh   = blockIdx.y;
  const int b    = bh >> 4;
  const int head = bh & 15;
  const int q0   = (blockIdx.x * 8 + wave) * 16;

  // Q fragments (B-matrix of T = K x Q^T): fixed for the whole key loop.
  const _Float16* qrow = Q + ((long)b * NN + q0 + ln) * EE + head * 64;
  const v16h qf0 = load_fragB(qrow, 0, h);
  const v16h qf1 = load_fragB(qrow, 32, h);

  const _Float16* Kbase = Kc + (long)b * NN * EE + head * 64;
  const _Float16* Vbase = Vt + (long)(b * 16 + head) * 64 * NN;

  // Copy work split across 256 threads: 256 K-transfers + 256 V-transfers of 16B.
  const int krow = tid >> 3, kseg = tid & 7;   // K: row 0..31, 8 segs of 8 halves
  const int vrow = tid >> 2, vseg = tid & 3;   // V: row d 0..63, 4 segs of 8 halves

  auto issue_copies = [&](int buf, int jb) {
    const uint32_t lK = (uint32_t)(uintptr_t)(&smem[buf][0]);
    const uint32_t lV = (uint32_t)(uintptr_t)(&smem[buf][32 * 64]);
    async_copy_b128(lK + (uint32_t)(krow * 64 + kseg * 8) * 2,
                    Kbase + (long)(jb + krow) * EE + kseg * 8);
    async_copy_b128(lV + (uint32_t)(vrow * 32 + vseg * 8) * 2,
                    Vbase + (long)vrow * NN + jb + vseg * 8);
  };

  v8f oacc[4] = {};
  float mrun = -1e30f;
  float lrun = 0.0f;
  int cur = 0;

  issue_copies(0, 0);

  for (int jb = 0; jb < NN; jb += 32) {
    wait_async0();
    __syncthreads();                     // staged K/V for `cur` visible to all waves
    if (jb + 32 < NN) issue_copies(cur ^ 1, jb + 32);

    const _Float16* Ks = &smem[cur][0];        // [32][64]
    const _Float16* Vs = &smem[cur][32 * 64];  // [64][32]

    // S^T tiles for keys [jb, jb+16) and [jb+16, jb+32)
    v8f t1 = v8f{}, t2 = v8f{};
    t1 = WMMA_F16(load_fragA(Ks + ln * 64, 0, h), qf0, t1);
    t1 = WMMA_F16(load_fragA(Ks + ln * 64, 32, h), qf1, t1);
    t2 = WMMA_F16(load_fragA(Ks + (16 + ln) * 64, 0, h), qf0, t2);
    t2 = WMMA_F16(load_fragA(Ks + (16 + ln) * 64, 32, h), qf1, t2);

    // Streaming softmax (per query = per lane; other 16 values live in lane^16)
    float p[16];
#pragma unroll
    for (int v = 0; v < 8; ++v) { p[v] = t1[v] * scale; p[8 + v] = t2[v] * scale; }
    float mx = -1e30f;
#pragma unroll
    for (int i = 0; i < 16; ++i) mx = fmaxf(mx, p[i]);
    mx = fmaxf(mx, __shfl_xor(mx, 16, 32));
    const float mnew  = fmaxf(mrun, mx);
    const float alpha = __expf(mrun - mnew);
    float ps = 0.0f;
#pragma unroll
    for (int i = 0; i < 16; ++i) { p[i] = __expf(p[i] - mnew); ps += p[i]; }
    ps += __shfl_xor(ps, 16, 32);
    lrun = lrun * alpha + ps;
    mrun = mnew;

    // Lane-local repack: p[] already matches the P A-fragment element order.
    v16h pf;
#pragma unroll
    for (int i = 0; i < 16; ++i) pf[i] = (_Float16)p[i];

    // Broadcast per-row rescale factors into C-layout rows (row r = v + 8h).
    float arow[8];
#pragma unroll
    for (int v = 0; v < 8; ++v) arow[v] = __shfl(alpha, v + 8 * h, 32);

    // O[16x64] += P[16x32] x V[32x64] (V^T rows in LDS, contiguous in j)
#pragma unroll
    for (int t = 0; t < 4; ++t) {
      v16h vf = load_fragB(Vs + (t * 16 + ln) * 32, 0, h);
      v8f c;
#pragma unroll
      for (int v = 0; v < 8; ++v) c[v] = oacc[t][v] * arow[v];
      oacc[t] = WMMA_F16(pf, vf, c);
    }

    __syncthreads();                     // all waves done reading `cur`
    cur ^= 1;
  }

  // Normalize and store (row-major f16 [B,N,E] for the output projection).
  const float li = 1.0f / lrun;
  float linv[8];
#pragma unroll
  for (int v = 0; v < 8; ++v) linv[v] = __shfl(li, v + 8 * h, 32);
#pragma unroll
  for (int t = 0; t < 4; ++t) {
#pragma unroll
    for (int v = 0; v < 8; ++v) {
      const int i = q0 + v + 8 * h;
      O[((long)b * NN + i) * EE + head * 64 + t * 16 + ln] = (_Float16)(oacc[t][v] * linv[v]);
    }
  }
}

// ---------------------------------------------------------------------------
// Host-side orchestration
// ---------------------------------------------------------------------------
extern "C" void kernel_launch(void* const* d_in, const int* in_sizes, int n_in,
                              void* d_out, int out_size, void* d_ws, size_t ws_size,
                              hipStream_t stream) {
  (void)in_sizes; (void)n_in; (void)out_size; (void)ws_size;
  const float* x  = (const float*)d_in[0];
  // d_in[1] = mask (all true) — no-op in this configuration
  const float* Wq = (const float*)d_in[2]; const float* bq = (const float*)d_in[3];
  const float* Wk = (const float*)d_in[4]; const float* bk = (const float*)d_in[5];
  const float* Wv = (const float*)d_in[6]; const float* bv = (const float*)d_in[7];
  const float* Wo = (const float*)d_in[8]; const float* bo = (const float*)d_in[9];

  const int M = 4096, E = 1024;          // M = B*N
  const size_t MB = 1024 * 1024;
  char* ws = (char*)d_ws;
  _Float16* xh  = (_Float16*)(ws);            //  8 MB  x in f16
  _Float16* wqh = (_Float16*)(ws + 8  * MB);  //  2 MB
  _Float16* wkh = (_Float16*)(ws + 10 * MB);  //  2 MB
  _Float16* wvh = (_Float16*)(ws + 12 * MB);  //  2 MB
  _Float16* woh = (_Float16*)(ws + 14 * MB);  //  2 MB
  _Float16* Qh  = (_Float16*)(ws + 16 * MB);  //  8 MB  [B,N,E]
  _Float16* Kh  = (_Float16*)(ws + 24 * MB);  //  8 MB  [B,N,E]
  _Float16* Vth = (_Float16*)(ws + 32 * MB);  //  8 MB  [B,H,D,N]
  _Float16* Oh  = (_Float16*)(ws + 40 * MB);  //  8 MB  [B,N,E]

  const int nx = M * E, nw = E * E;
  cvt_f32_f16<<<(nx + 255) / 256, 256, 0, stream>>>(x,  xh,  nx);
  cvt_f32_f16<<<(nw + 255) / 256, 256, 0, stream>>>(Wq, wqh, nw);
  cvt_f32_f16<<<(nw + 255) / 256, 256, 0, stream>>>(Wk, wkh, nw);
  cvt_f32_f16<<<(nw + 255) / 256, 256, 0, stream>>>(Wv, wvh, nw);
  cvt_f32_f16<<<(nw + 255) / 256, 256, 0, stream>>>(Wo, woh, nw);

  dim3 gg(M / 32, 4), gb(128);           // wave tile 32x64; 4 waves x 64 cols per y-block
  gemm_wmma<<<gg, gb, 0, stream>>>(xh, wqh, bq, Qh,  M, E, 0);
  gemm_wmma<<<gg, gb, 0, stream>>>(xh, wkh, bk, Kh,  M, E, 0);
  gemm_wmma<<<gg, gb, 0, stream>>>(xh, wvh, bv, Vth, M, E, 1);

  attn_flash<<<dim3(16, 32), 256, 0, stream>>>(Qh, Kh, Vth, Oh);

  gemm_wmma<<<gg, gb, 0, stream>>>(Oh, woh, bo, d_out, M, E, 2);
}
